// SpatialAttentionBlock_89309549953899
// MI455X (gfx1250) — matrix operand order repaired and checked by reading, not compiled, not run on hardware
//
#include <hip/hip_runtime.h>
#include <hip/hip_bf16.h>
#include <math.h>

// ---------------------------------------------------------------------------
// Problem constants (from reference)
// ---------------------------------------------------------------------------
#define BATCH   2
#define DIM     1024
#define HEADS   16
#define HEAD_DIM 64
#define SEQ     2048
#define HIDDEN  4096
#define BN      (BATCH * SEQ)          // 4096 rows of activations
#define QKSCALE 0.125f                 // HEAD_DIM^-0.5
#define LN_EPS  1e-5f

// ---------------------------------------------------------------------------
// WMMA vector types (CDNA5 gfx1250, wave32)
// ---------------------------------------------------------------------------
typedef __attribute__((ext_vector_type(16))) __bf16 v16bf;
typedef __attribute__((ext_vector_type(8)))  __bf16 v8bf;
typedef __attribute__((ext_vector_type(8)))  float  v8f;

union BF16Frag { v16bf v; v8bf h8[2]; };

// ---------------------------------------------------------------------------
// Small elementwise kernels
// ---------------------------------------------------------------------------
__global__ void f32_to_bf16_kernel(const float* __restrict__ s,
                                   __bf16* __restrict__ d, int n) {
    int i = blockIdx.x * blockDim.x + threadIdx.x;
    if (i < n) d[i] = (__bf16)s[i];
}

// xt[b][n][c] = x[b][c][n]  (output-indexed; reads hit L2 repeatedly)
__global__ void transpose_x_kernel(const float* __restrict__ x,
                                   __bf16* __restrict__ xt) {
    size_t tid = (size_t)blockIdx.x * blockDim.x + threadIdx.x;
    if (tid >= (size_t)BN * DIM) return;
    int c = (int)(tid & (DIM - 1));
    size_t rn = tid >> 10;                 // b*SEQ + n
    int n = (int)(rn & (SEQ - 1));
    int b = (int)(rn >> 11);
    xt[tid] = (__bf16)x[((size_t)b * DIM + c) * SEQ + n];
}

// qkv (BN x 3072) -> q_scaled (B,H,N,Dh), k (B,H,N,Dh), vT (B,H,Dh,N)
__global__ void reshape_qkv_kernel(const __bf16* __restrict__ qkv,
                                   __bf16* __restrict__ qs,
                                   __bf16* __restrict__ kk,
                                   __bf16* __restrict__ vt) {
    size_t tid = (size_t)blockIdx.x * blockDim.x + threadIdx.x;
    if (tid >= (size_t)BN * DIM) return;
    int c = (int)(tid & (DIM - 1));        // h*64 + d
    size_t rn = tid >> 10;                 // b*SEQ + n
    int n = (int)(rn & (SEQ - 1));
    int b = (int)(rn >> 11);
    int h = c >> 6, d = c & 63;
    const __bf16* base = qkv + rn * (3 * DIM);
    float qv = (float)base[c];
    float kv = (float)base[DIM + c];
    float vv = (float)base[2 * DIM + c];
    size_t bh = (size_t)(b * HEADS + h);
    qs[(bh * SEQ + n) * HEAD_DIM + d] = (__bf16)(qv * QKSCALE);
    kk[(bh * SEQ + n) * HEAD_DIM + d] = (__bf16)kv;
    vt[(bh * HEAD_DIM + d) * SEQ + n] = (__bf16)vv;
}

// LayerNorm over HIDDEN per row, write bf16
__global__ __launch_bounds__(256)
void layernorm_kernel(const float* __restrict__ m,
                      const float* __restrict__ g,
                      const float* __restrict__ bta,
                      __bf16* __restrict__ out) {
    __shared__ float red[256];
    const int row = blockIdx.x;
    const float* p = m + (size_t)row * HIDDEN;
    float s = 0.f;
    for (int c = threadIdx.x; c < HIDDEN; c += 256) s += p[c];
    red[threadIdx.x] = s; __syncthreads();
    for (int st = 128; st > 0; st >>= 1) {
        if (threadIdx.x < st) red[threadIdx.x] += red[threadIdx.x + st];
        __syncthreads();
    }
    float mu = red[0] * (1.f / HIDDEN);
    __syncthreads();
    float v = 0.f;
    for (int c = threadIdx.x; c < HIDDEN; c += 256) {
        float dlt = p[c] - mu; v += dlt * dlt;
    }
    red[threadIdx.x] = v; __syncthreads();
    for (int st = 128; st > 0; st >>= 1) {
        if (threadIdx.x < st) red[threadIdx.x] += red[threadIdx.x + st];
        __syncthreads();
    }
    float rstd = rsqrtf(red[0] * (1.f / HIDDEN) + LN_EPS);
    for (int c = threadIdx.x; c < HIDDEN; c += 256)
        out[(size_t)row * HIDDEN + c] = (__bf16)((p[c] - mu) * rstd * g[c] + bta[c]);
}

// ---------------------------------------------------------------------------
// Tiled bf16 WMMA GEMM with double-buffered LDS staging:
//   C[M,N] = A[M,K] @ W[N,K]^T + bias, fused epilogues.
// Block = 256 threads = 8 waves (2 row-groups x 4 col-groups), tile 128x128.
// LDS: A/B tiles 128x32 bf16, row stride padded to 40 halves (80B) so the
// fragment ds_load_b128s from 16 consecutive rows cover all 64 banks
// (gcd(20 dwords, 64 banks) = 4 -> 16 disjoint 4-dword groups, conflict-free).
// Per K-step a block moves 16KB from L2 for 1.05 MFLOP (~64 flops/byte).
// MODE 0: qkv   -> bf16 Cb
// MODE 1: proj  -> t = 2*(acc+bias); bf16 Cb and fp32 Cf (residual h)
// MODE 2: fc1   -> exact GELU; fp32 Cf
// MODE 3: fc2   -> acc+bias+Res; scatter to (B,C,N) fp32 output Cf
// ---------------------------------------------------------------------------
#define LDSTRIDE 40                    // halves per LDS tile row (32 + 8 pad)
#define LDBUF    (128 * LDSTRIDE)

template <int MODE>
__global__ __launch_bounds__(256)
void gemm_bf16_kernel(const __bf16* __restrict__ A,
                      const __bf16* __restrict__ W,
                      const float* __restrict__ bias,
                      __bf16* __restrict__ Cb,
                      float* __restrict__ Cf,
                      const float* __restrict__ Res,
                      int M, int Nout, int K) {
    __shared__ __bf16 sA[2 * LDBUF];
    __shared__ __bf16 sB[2 * LDBUF];

    const int lane = threadIdx.x & 31;
    const int wave = threadIdx.x >> 5;
    const int wr = wave >> 2;                       // 0..1 row group
    const int wc = wave & 3;                        // 0..3 col group
    const int row0 = blockIdx.y * 128 + wr * 64;
    const int col0 = blockIdx.x * 128 + wc * 32;
    const int r = lane & 15, hi = lane >> 4;

    // Cooperative loader: each thread owns 32B (16 halves) of each tile.
    const int ldr = threadIdx.x >> 1;               // 0..127
    const int ldk = (threadIdx.x & 1) << 4;         // 0 or 16
    const __bf16* gA = A + (size_t)(blockIdx.y * 128 + ldr) * K + ldk;
    const __bf16* gB = W + (size_t)(blockIdx.x * 128 + ldr) * K + ldk;
    const int sOff = ldr * LDSTRIDE + ldk;

    v8f acc[4][2] = {};

    // Preload tile 0 into buffer 0
    {
        v8bf a0 = *(const v8bf*)gA, a1 = *(const v8bf*)(gA + 8);
        v8bf b0 = *(const v8bf*)gB, b1 = *(const v8bf*)(gB + 8);
        *(v8bf*)&sA[sOff] = a0; *(v8bf*)&sA[sOff + 8] = a1;
        *(v8bf*)&sB[sOff] = b0; *(v8bf*)&sB[sOff + 8] = b1;
    }
    __syncthreads();

    const int nk = K >> 5;
    v8bf ra0, ra1, rb0, rb1;
    for (int kt = 0; kt < nk; ++kt) {
        const int cur = (kt & 1) * LDBUF;
        const int nxt = ((kt + 1) & 1) * LDBUF;
        const bool hasNext = (kt + 1) < nk;

        if (hasNext) {                              // issue next-tile loads first
            const __bf16* pa = gA + (size_t)(kt + 1) * 32;
            const __bf16* pb = gB + (size_t)(kt + 1) * 32;
            ra0 = *(const v8bf*)pa; ra1 = *(const v8bf*)(pa + 8);
            rb0 = *(const v8bf*)pb; rb1 = *(const v8bf*)(pb + 8);
            if (kt + 3 < nk) {                      // CDNA5 global_prefetch_b8
                __builtin_prefetch(gA + (size_t)(kt + 3) * 32, 0, 3);
                __builtin_prefetch(gB + (size_t)(kt + 3) * 32, 0, 3);
            }
        }

        // Compute on current LDS tiles (hides the global-load latency)
        BF16Frag bfr[2];
#pragma unroll
        for (int ct = 0; ct < 2; ++ct) {
            const __bf16* p = &sB[cur + (wc * 32 + ct * 16 + r) * LDSTRIDE + hi * 16];
            bfr[ct].h8[0] = *(const v8bf*)p;        // K = 16*hi + 0..7
            bfr[ct].h8[1] = *(const v8bf*)(p + 8);  // K = 16*hi + 8..15
        }
#pragma unroll
        for (int rt = 0; rt < 4; ++rt) {
            BF16Frag af;
            const __bf16* p = &sA[cur + (wr * 64 + rt * 16 + r) * LDSTRIDE + hi * 16];
            af.h8[0] = *(const v8bf*)p;             // K = 16*hi + 0..7
            af.h8[1] = *(const v8bf*)(p + 8);       // K = 16*hi + 8..15
#pragma unroll
            for (int ct = 0; ct < 2; ++ct) {
                acc[rt][ct] = __builtin_amdgcn_wmma_f32_16x16x32_bf16(
                    false, af.v, false, bfr[ct].v, (short)0, acc[rt][ct],
                    false, false);
            }
        }

        if (hasNext) {                              // drain regs into other buffer
            *(v8bf*)&sA[nxt + sOff] = ra0; *(v8bf*)&sA[nxt + sOff + 8] = ra1;
            *(v8bf*)&sB[nxt + sOff] = rb0; *(v8bf*)&sB[nxt + sOff + 8] = rb1;
        }
        __syncthreads();
    }

    // Epilogue: C layout — lane holds (row = rt*16 + v + 8*hi, col = ct*16 + r)
#pragma unroll
    for (int rt = 0; rt < 4; ++rt) {
#pragma unroll
        for (int ct = 0; ct < 2; ++ct) {
#pragma unroll
            for (int v = 0; v < 8; ++v) {
                int row = row0 + rt * 16 + v + 8 * hi;
                int col = col0 + ct * 16 + r;
                float t = acc[rt][ct][v] + bias[col];
                if (MODE == 0) {
                    Cb[(size_t)row * Nout + col] = (__bf16)t;
                } else if (MODE == 1) {
                    float t2 = 2.0f * t;                 // out + ls1(out), ls1=Id
                    Cb[(size_t)row * Nout + col] = (__bf16)t2;
                    Cf[(size_t)row * Nout + col] = t2;
                } else if (MODE == 2) {
                    float ge = 0.5f * t * (1.0f + erff(t * 0.70710678118654752f));
                    Cf[(size_t)row * Nout + col] = ge;
                } else { // MODE 3: y = h + m, scatter to (B, C, N)
                    float y = t + Res[(size_t)row * DIM + col];
                    int b = row >> 11, n = row & (SEQ - 1);
                    Cf[((size_t)b * DIM + col) * SEQ + n] = y;
                }
            }
        }
    }
}

// ---------------------------------------------------------------------------
// Flash attention with softmax-one, computed as S^T = K @ Q^T so the score
// C-layout (q-row == lane&15) feeds the P A-fragment of O = P@V with zero
// cross-lane transposes. Block = 4 waves; wave owns 16 q-rows of one (b,h).
// ---------------------------------------------------------------------------
__global__ __launch_bounds__(128)
void attention_kernel(const __bf16* __restrict__ qs,
                      const __bf16* __restrict__ kk,
                      const __bf16* __restrict__ vt,
                      const int* __restrict__ mask,
                      __bf16* __restrict__ am) {
    const int lane = threadIdx.x & 31;
    const int wave = threadIdx.x >> 5;
    const int bh = blockIdx.y;                       // b*HEADS + h
    const int q0 = blockIdx.x * 64 + wave * 16;
    const int b = bh >> 4, h = bh & 15;
    const int r = lane & 15, hi = lane >> 4;

    // Q as B-fragment (column = q-row, K = d): loop-invariant, 2 k-steps
    const __bf16* qrow = qs + ((size_t)bh * SEQ + (q0 + r)) * HEAD_DIM;
    BF16Frag qf[2];
#pragma unroll
    for (int t = 0; t < 2; ++t) {
        const __bf16* p = qrow + t * 32 + hi * 8;
        qf[t].h8[0] = *(const v8bf*)p;
        qf[t].h8[1] = *(const v8bf*)(p + 16);
    }

    const bool rowOk = mask[(size_t)bh * SEQ + (q0 + r)] != 0;
    float m_run = -INFINITY, s_run = 0.f;
    v8f o[4] = {};

    const __bf16* kbase = kk + (size_t)bh * SEQ * HEAD_DIM;
    const __bf16* vbase = vt + (size_t)bh * HEAD_DIM * SEQ;

    for (int j = 0; j < SEQ; j += 32) {
        v8f st0 = {}, st1 = {};                       // S^T tiles (keys x qrows)
#pragma unroll
        for (int t = 0; t < 2; ++t) {                 // K-dim over d (0,32)
            BF16Frag a0, a1;
            const __bf16* p0 = kbase + (size_t)(j + r) * HEAD_DIM + t * 32 + hi * 8;
            a0.h8[0] = *(const v8bf*)p0;
            a0.h8[1] = *(const v8bf*)(p0 + 16);
            const __bf16* p1 = kbase + (size_t)(j + 16 + r) * HEAD_DIM + t * 32 + hi * 8;
            a1.h8[0] = *(const v8bf*)p1;
            a1.h8[1] = *(const v8bf*)(p1 + 16);
            st0 = __builtin_amdgcn_wmma_f32_16x16x32_bf16(
                false, a0.v, false, qf[t].v, (short)0, st0, false, false);
            st1 = __builtin_amdgcn_wmma_f32_16x16x32_bf16(
                false, a1.v, false, qf[t].v, (short)0, st1, false, false);
        }

        // Online softmax-one stats; each lane's values belong to q-row (lane&15)
        float p0a[8], p1a[8];
        float mt = -INFINITY;
#pragma unroll
        for (int v = 0; v < 8; ++v) {
            float x0 = rowOk ? st0[v] : -1e9f;
            float x1 = rowOk ? st1[v] : -1e9f;
            p0a[v] = x0; p1a[v] = x1;
            mt = fmaxf(mt, fmaxf(x0, x1));
        }
        mt = fmaxf(mt, __shfl_xor(mt, 16, 32));
        float m_new = fmaxf(m_run, mt);
        float alpha = __expf(m_run - m_new);
        float rs = 0.f;
#pragma unroll
        for (int v = 0; v < 8; ++v) {
            p0a[v] = __expf(p0a[v] - m_new);
            p1a[v] = __expf(p1a[v] - m_new);
            rs += p0a[v] + p1a[v];
        }
        rs += __shfl_xor(rs, 16, 32);
        s_run = s_run * alpha + rs;
        m_run = m_new;

        // Rescale O (O rows live across registers: row = v + 8*hi)
        float al[8];
#pragma unroll
        for (int v = 0; v < 8; ++v) al[v] = __shfl(alpha, v + 8 * hi, 32);
#pragma unroll
        for (int g = 0; g < 4; ++g)
#pragma unroll
            for (int v = 0; v < 8; ++v) o[g][v] *= al[v];

        // P A-fragment directly from S^T registers (no shuffles needed)
        v16bf pa;
#pragma unroll
        for (int v = 0; v < 8; ++v) {
            pa[v]     = (__bf16)p0a[v];               // K = 8*hi + v
            pa[8 + v] = (__bf16)p1a[v];               // K = 16 + 8*hi + v
        }

        // O += P @ V  (V via vT: contiguous B-fragment loads)
#pragma unroll
        for (int g = 0; g < 4; ++g) {
            BF16Frag vf;
            const __bf16* vp = vbase + (size_t)(g * 16 + r) * SEQ + j + hi * 8;
            vf.h8[0] = *(const v8bf*)vp;
            vf.h8[1] = *(const v8bf*)(vp + 16);
            o[g] = __builtin_amdgcn_wmma_f32_16x16x32_bf16(
                false, pa, false, vf.v, (short)0, o[g], false, false);
        }
    }

    // softmax-one: denominator = 1 + sum(exp(x - m))
    float inv = 1.0f / (1.0f + s_run);
    float iv[8];
#pragma unroll
    for (int v = 0; v < 8; ++v) iv[v] = __shfl(inv, v + 8 * hi, 32);

    // Store with the reference's scrambled (0,3,1,2)-merge folded in
#pragma unroll
    for (int g = 0; g < 4; ++g) {
#pragma unroll
        for (int v = 0; v < 8; ++v) {
            float val = o[g][v] * iv[v];
            int n = q0 + v + 8 * hi;
            int d = g * 16 + r;
            size_t idx = (size_t)d * (HEADS * SEQ) + (size_t)h * SEQ + n;
            int rr = (int)(idx >> 10);                // / DIM
            int cc = (int)(idx & (DIM - 1));
            am[((size_t)b * SEQ + rr) * DIM + cc] = (__bf16)val;
        }
    }
}

// ---------------------------------------------------------------------------
// Host launcher. Workspace layout (~218 MB, all offsets 256B-aligned).
// ---------------------------------------------------------------------------
extern "C" void kernel_launch(void* const* d_in, const int* in_sizes, int n_in,
                              void* d_out, int out_size, void* d_ws, size_t ws_size,
                              hipStream_t stream) {
    (void)in_sizes; (void)n_in; (void)out_size; (void)ws_size;

    const float* x      = (const float*)d_in[0];
    const int*   mask   = (const int*)  d_in[1];
    const float* qkv_w  = (const float*)d_in[2];
    const float* qkv_b  = (const float*)d_in[3];
    const float* proj_w = (const float*)d_in[4];
    const float* proj_b = (const float*)d_in[5];
    const float* fc1_w  = (const float*)d_in[6];
    const float* fc1_b  = (const float*)d_in[7];
    const float* ln_g   = (const float*)d_in[8];
    const float* ln_b   = (const float*)d_in[9];
    const float* fc2_w  = (const float*)d_in[10];
    const float* fc2_b  = (const float*)d_in[11];
    float* out = (float*)d_out;

    char* ws = (char*)d_ws;
    size_t off = 0;
    auto carve = [&](size_t bytes) { char* p = ws + off; off += (bytes + 255) & ~(size_t)255; return p; };

    __bf16* xt_bf  = (__bf16*)carve((size_t)BN * DIM * 2);            // 8 MB
    __bf16* wq_bf  = (__bf16*)carve((size_t)3 * DIM * DIM * 2);       // 6 MB
    __bf16* wp_bf  = (__bf16*)carve((size_t)DIM * DIM * 2);           // 2 MB
    __bf16* w1_bf  = (__bf16*)carve((size_t)HIDDEN * DIM * 2);        // 8 MB
    __bf16* w2_bf  = (__bf16*)carve((size_t)DIM * HIDDEN * 2);        // 8 MB
    __bf16* qkv_bf = (__bf16*)carve((size_t)BN * 3 * DIM * 2);        // 24 MB
    __bf16* qs_bf  = (__bf16*)carve((size_t)BN * DIM * 2);            // 8 MB
    __bf16* k_bf   = (__bf16*)carve((size_t)BN * DIM * 2);            // 8 MB
    __bf16* vt_bf  = (__bf16*)carve((size_t)BN * DIM * 2);            // 8 MB
    __bf16* am_bf  = (__bf16*)carve((size_t)BN * DIM * 2);            // 8 MB
    float*  h_f32  = (float*) carve((size_t)BN * DIM * 4);            // 16 MB
    __bf16* h_bf   = (__bf16*)carve((size_t)BN * DIM * 2);            // 8 MB
    float*  m_f32  = (float*) carve((size_t)BN * HIDDEN * 4);         // 64 MB
    __bf16* mn_bf  = (__bf16*)carve((size_t)BN * HIDDEN * 2);         // 32 MB

    const int cvt_block = 256;
    auto cvt = [&](const float* s, __bf16* d, int n) {
        f32_to_bf16_kernel<<<(n + cvt_block - 1) / cvt_block, cvt_block, 0, stream>>>(s, d, n);
    };
    cvt(qkv_w,  wq_bf, 3 * DIM * DIM);
    cvt(proj_w, wp_bf, DIM * DIM);
    cvt(fc1_w,  w1_bf, HIDDEN * DIM);
    cvt(fc2_w,  w2_bf, DIM * HIDDEN);

    {   // x (B,C,N) -> xt bf16 (B,N,C)
        size_t n = (size_t)BN * DIM;
        transpose_x_kernel<<<(unsigned)((n + 255) / 256), 256, 0, stream>>>(x, xt_bf);
    }

    // qkv = xt @ qkv_w^T + b : (4096 x 1024) x (3072 x 1024)^T
    gemm_bf16_kernel<0><<<dim3(3 * DIM / 128, BN / 128), 256, 0, stream>>>(
        xt_bf, wq_bf, qkv_b, qkv_bf, nullptr, nullptr, BN, 3 * DIM, DIM);

    {   // reshape to per-head q (scaled), k, vT
        size_t n = (size_t)BN * DIM;
        reshape_qkv_kernel<<<(unsigned)((n + 255) / 256), 256, 0, stream>>>(
            qkv_bf, qs_bf, k_bf, vt_bf);
    }

    // flash attention with softmax-one + scrambled merge
    attention_kernel<<<dim3(SEQ / 64, BATCH * HEADS), 128, 0, stream>>>(
        qs_bf, k_bf, vt_bf, mask, am_bf);

    // h = 2*(attn_merged @ proj_w^T + b)  -> bf16 (for fc1) and fp32 (residual)
    gemm_bf16_kernel<1><<<dim3(DIM / 128, BN / 128), 256, 0, stream>>>(
        am_bf, wp_bf, proj_b, h_bf, h_f32, nullptr, BN, DIM, DIM);

    // m = gelu(h @ fc1_w^T + b) -> fp32
    gemm_bf16_kernel<2><<<dim3(HIDDEN / 128, BN / 128), 256, 0, stream>>>(
        h_bf, w1_bf, fc1_b, nullptr, m_f32, nullptr, BN, HIDDEN, DIM);

    // LayerNorm over HIDDEN
    layernorm_kernel<<<BN, 256, 0, stream>>>(m_f32, ln_g, ln_b, mn_bf);

    // y = h + (mn @ fc2_w^T + b), scattered to (B, C, N)
    gemm_bf16_kernel<3><<<dim3(DIM / 128, BN / 128), 256, 0, stream>>>(
        mn_bf, w2_bf, fc2_b, nullptr, out, h_f32, BN, DIM, HIDDEN);
}
